// PreLNTransformerLayer_65025804861862
// MI455X (gfx1250) — matrix operand order repaired and checked by reading, not compiled, not run on hardware
//
#include <hip/hip_runtime.h>
#include <hip/hip_bf16.h>
#include <math.h>

// ---------------------------------------------------------------------------
// Types for CDNA5 WMMA
// ---------------------------------------------------------------------------
typedef __attribute__((ext_vector_type(16))) __bf16 v16bf;
typedef __attribute__((ext_vector_type(8)))  float  v8f;

union FragB16 { v16bf v; unsigned u[8]; };

__device__ __forceinline__ v8f wmma_bf16(v16bf a, v16bf b, v8f c) {
  // D = A(16x32 bf16) * B(32x16 bf16) + C(16x16 f32)
  return __builtin_amdgcn_wmma_f32_16x16x32_bf16(
      /*neg_a=*/false, a, /*neg_b=*/false, b,
      /*c_mod=*/(short)0, c, /*reuse_a=*/false, /*reuse_b=*/false);
}

// A-matrix fragment (16x32, bf16), source row-major with leading dim `ld`.
// ISA layout: lane holds M=lane&15; k = 2*(v&3) + 8*half + 16*(v>>2)
__device__ __forceinline__ v16bf load_frag_a(const __bf16* base, int ld,
                                             int hlf, int lr) {
  FragB16 f;
#pragma unroll
  for (int v = 0; v < 8; ++v) {
    int k = 2 * (v & 3) + 8 * hlf + 16 * (v >> 2);
    f.u[v] = *(const unsigned*)(base + (size_t)lr * ld + k);
  }
  return f.v;
}

// B-matrix fragment (32x16, bf16) where B[k][n] = src[n*ld + k]
// (source stores each n-row contiguous in k). lane: n = lane&15, k = 2*(v&3) + 8*half + 16*(v>>2)
__device__ __forceinline__ v16bf load_frag_b_k(const __bf16* base, int ld,
                                               int hlf, int lr) {
  FragB16 f;
#pragma unroll
  for (int v = 0; v < 8; ++v) {
    int k = 2 * (v & 3) + 8 * hlf + 16 * (v >> 2);
    f.u[v] = *(const unsigned*)(base + (size_t)lr * ld + k);
  }
  return f.v;
}

// CDNA5 async copy: global -> LDS, 16 bytes per lane, tracked by ASYNCcnt.
__device__ __forceinline__ void async_g2l_b128(unsigned lds_addr,
                                               const void* gptr) {
  asm volatile("global_load_async_to_lds_b128 %0, %1, off"
               :: "v"(lds_addr), "v"((unsigned long long)(size_t)gptr)
               : "memory");
}
__device__ __forceinline__ void wait_asynccnt0() {
  asm volatile("s_wait_asynccnt 0x0" ::: "memory");
}
// Compiler-only ordering fence for intra-wave LDS exchange (hardware keeps
// LDS in-order per wave; we only need to stop compiler reordering).
__device__ __forceinline__ void lds_fence() {
  asm volatile("" ::: "memory");
}

// ---------------------------------------------------------------------------
// fp32 -> bf16 conversion (weights)
// ---------------------------------------------------------------------------
__global__ void f2bf_kernel(const float* __restrict__ in,
                            __bf16* __restrict__ out, size_t n) {
  for (size_t i = (size_t)blockIdx.x * blockDim.x + threadIdx.x; i < n;
       i += (size_t)gridDim.x * blockDim.x)
    out[i] = (__bf16)in[i];
}

// ---------------------------------------------------------------------------
// LayerNorm: fp32 in -> bf16 out (one block per row)
// ---------------------------------------------------------------------------
__global__ void ln_kernel(const float* __restrict__ x,
                          const float* __restrict__ g,
                          const float* __restrict__ bta,
                          __bf16* __restrict__ y, int d) {
  __shared__ float s1[256], s2[256];
  const int row = blockIdx.x, tid = threadIdx.x;
  const float* xr = x + (size_t)row * d;
  float a = 0.f, b = 0.f;
  for (int i = tid; i < d; i += 256) { float v = xr[i]; a += v; b += v * v; }
  s1[tid] = a; s2[tid] = b;
  __syncthreads();
  for (int s = 128; s > 0; s >>= 1) {
    if (tid < s) { s1[tid] += s1[tid + s]; s2[tid] += s2[tid + s]; }
    __syncthreads();
  }
  const float mu = s1[0] / d;
  const float var = s2[0] / d - mu * mu;
  const float rinv = rsqrtf(var + 1e-5f);
  __bf16* yr = y + (size_t)row * d;
  for (int i = tid; i < d; i += 256)
    yr[i] = (__bf16)((xr[i] - mu) * rinv * g[i] + bta[i]);
}

// ---------------------------------------------------------------------------
// Tiled bf16 GEMM:  out[M,N] = A[M,K] @ W[N,K]^T + bias  (+GELU) (+resid)
// Block tile 128x128, BK=32. 256 threads = 8 waves (4 in M x 2 in N),
// each wave computes 32x64 via 2x4 v_wmma_f32_16x16x32_bf16 tiles.
// Staging: double-buffered LDS filled with global_load_async_to_lds_b128,
// software-pipelined (next tile's async copies overlap current compute).
// ---------------------------------------------------------------------------
#define GLD 40   // padded LDS leading dim (elements): 80B rows, 16B-aligned

__global__ __launch_bounds__(256)
void gemm_kernel(const __bf16* __restrict__ A, const __bf16* __restrict__ W,
                 const float* __restrict__ bias, const float* __restrict__ resid,
                 float* __restrict__ outF, __bf16* __restrict__ outB,
                 int M, int N, int K, int do_gelu) {
  __shared__ __bf16 As[2][128 * GLD];
  __shared__ __bf16 Ws[2][128 * GLD];

  const int tid = threadIdx.x;
  const int w = tid >> 5, lane = tid & 31;
  const int hlf = lane >> 4, lr = lane & 15;
  const int wm = w >> 1, wn = w & 1;        // 4 x 2 waves
  const int bm = blockIdx.x * 128, bn = blockIdx.y * 128;
  const int mt = wm * 32, nt = wn * 64;

  // Staging: tile is 128 rows x 4 16B-chunks = 512 chunks; 2 each of A and W.
  const int c0 = tid, c1 = 256 + tid;
  const int ra0 = c0 >> 2, pa0 = c0 & 3;
  const int ra1 = c1 >> 2, pa1 = c1 & 3;

  auto issue = [&](int k0, int buf) {
    async_g2l_b128((unsigned)(size_t)&As[buf][ra0 * GLD + pa0 * 8],
                   &A[(size_t)(bm + ra0) * K + k0 + pa0 * 8]);
    async_g2l_b128((unsigned)(size_t)&As[buf][ra1 * GLD + pa1 * 8],
                   &A[(size_t)(bm + ra1) * K + k0 + pa1 * 8]);
    async_g2l_b128((unsigned)(size_t)&Ws[buf][ra0 * GLD + pa0 * 8],
                   &W[(size_t)(bn + ra0) * K + k0 + pa0 * 8]);
    async_g2l_b128((unsigned)(size_t)&Ws[buf][ra1 * GLD + pa1 * 8],
                   &W[(size_t)(bn + ra1) * K + k0 + pa1 * 8]);
  };

  v8f acc[2][4] = {};
  int cur = 0;
  issue(0, 0);

  for (int k0 = 0; k0 < K; k0 += 32) {
    wait_asynccnt0();        // my async writes to buf[cur] done
    __syncthreads();         // everyone's writes visible / reads of prev done
    if (k0 + 32 < K) issue(k0 + 32, cur ^ 1);   // overlap with compute below

    v16bf af[2], bw[4];
#pragma unroll
    for (int i = 0; i < 2; ++i)
      af[i] = load_frag_a(&As[cur][(mt + i * 16) * GLD], GLD, hlf, lr);
#pragma unroll
    for (int j = 0; j < 4; ++j)
      bw[j] = load_frag_b_k(&Ws[cur][(nt + j * 16) * GLD], GLD, hlf, lr);
#pragma unroll
    for (int i = 0; i < 2; ++i)
#pragma unroll
      for (int j = 0; j < 4; ++j)
        acc[i][j] = wmma_bf16(af[i], bw[j], acc[i][j]);
    cur ^= 1;
  }

  // epilogue: C layout is m = r + 8*half, n = lane&15
#pragma unroll
  for (int i = 0; i < 2; ++i)
#pragma unroll
    for (int j = 0; j < 4; ++j)
#pragma unroll
      for (int r = 0; r < 8; ++r) {
        int gm = bm + mt + i * 16 + r + 8 * hlf;
        int gn = bn + nt + j * 16 + lr;
        float v = acc[i][j][r] + bias[gn];
        if (do_gelu) v = 0.5f * v * (1.0f + erff(v * 0.70710678118654752f));
        if (resid) v += resid[(size_t)gm * N + gn];
        if (outF) outF[(size_t)gm * N + gn] = v;
        if (outB) outB[(size_t)gm * N + gn] = (__bf16)v;
      }
}

// ---------------------------------------------------------------------------
// qkv [B,S,3,H,hd] (bf16) -> Q,K [B,H,S,hd] ; V transposed [B,H,hd,S]
// ---------------------------------------------------------------------------
__global__ void qkv_split_kernel(const __bf16* __restrict__ qkv,
                                 __bf16* __restrict__ Q, __bf16* __restrict__ K,
                                 __bf16* __restrict__ Vt) {
  const size_t total = (size_t)2 * 2048 * 1024;
  for (size_t idx = (size_t)blockIdx.x * blockDim.x + threadIdx.x; idx < total;
       idx += (size_t)gridDim.x * blockDim.x) {
    int e = idx & 63;
    int h = (idx >> 6) & 15;
    int s = (idx >> 10) & 2047;
    int b = (int)(idx >> 21);
    size_t src = ((size_t)(b * 2048 + s)) * 3072 + h * 64 + e;
    size_t bh = (size_t)(b * 16 + h) * 2048 * 64;
    Q[bh + (size_t)s * 64 + e] = qkv[src];
    K[bh + (size_t)s * 64 + e] = qkv[src + 1024];
    Vt[bh + (size_t)e * 2048 + s] = qkv[src + 2048];
  }
}

// ---------------------------------------------------------------------------
// Flash attention with additive bias.
// One block owns (b, h, 128 consecutive queries); its 8 waves each own a
// 16-query tile. K/V 32-key tiles are staged ONCE per block in LDS through a
// double-buffered global_load_async_to_lds_b128 pipeline (8x less K/V
// traffic than per-wave streaming). Per-key-step: 4 score WMMAs, online
// softmax through per-wave LDS (intra-wave => compiler fences only),
// 4 P@V WMMAs. One block barrier per key-step.
// ---------------------------------------------------------------------------
#define KLD 72   // K tile leading dim (elems): 144B rows, 16B-aligned
#define VLD 40   // V tile leading dim (elems):  80B rows, 16B-aligned

__global__ __launch_bounds__(256)
void attn_kernel(const __bf16* __restrict__ Q, const __bf16* __restrict__ Km,
                 const __bf16* __restrict__ Vt, const float* __restrict__ bias,
                 __bf16* __restrict__ out) {
  __shared__ __bf16 Ks[2][32 * KLD];   // [key][hd]
  __shared__ __bf16 Vs[2][64 * VLD];   // [hd][key]
  __shared__ float  sS[8][16 * 33];    // per-wave score tile 16x32 f32
  __shared__ __bf16 sP[8][16 * 34];    // per-wave prob tile 16x32 bf16
  __shared__ float  sAl[8][16];        // per-row alpha
  __shared__ float  sL[8][16];         // per-row softmax denominator

  const int tid = threadIdx.x;
  const int w = tid >> 5, lane = tid & 31;
  const int hlf = lane >> 4, lr = lane & 15;
  const int blk = blockIdx.x;
  const int qb = (blk & 15) * 128;      // query block
  const int h = (blk >> 4) & 15;
  const int b = blk >> 8;
  const int q0 = qb + w * 16;           // this wave's 16 query rows

  const size_t bh = (size_t)(b * 16 + h) * 2048 * 64;
  const __bf16* Qp = Q + bh + (size_t)q0 * 64;
  const float* Bp = bias + (size_t)h * 2048 * 2048 + (size_t)q0 * 2048;

  // Staging shares: K tile 32 rows x 8 chunks, V tile 64 rows x 4 chunks.
  const int krow = tid >> 3, kcp = tid & 7;
  const int vrow = tid >> 2, vcp = tid & 3;
  auto issueKV = [&](int kt, int buf) {
    async_g2l_b128((unsigned)(size_t)&Ks[buf][krow * KLD + kcp * 8],
                   &Km[bh + (size_t)(kt + krow) * 64 + kcp * 8]);
    async_g2l_b128((unsigned)(size_t)&Vs[buf][vrow * VLD + vcp * 8],
                   &Vt[bh + (size_t)vrow * 2048 + kt + vcp * 8]);
  };

  const v16bf a0 = load_frag_a(Qp, 64, hlf, lr);        // hd 0..31
  const v16bf a1 = load_frag_a(Qp + 32, 64, hlf, lr);   // hd 32..63

  v8f acc[4] = {};
  float mrow = -1e30f, lrow = 0.f;    // stats for row = lane&15
  int cur = 0;
  issueKV(0, 0);

  for (int kt = 0; kt < 2048; kt += 32) {
    wait_asynccnt0();
    __syncthreads();                     // K/V tile[cur] ready block-wide
    if (kt + 32 < 2048) issueKV(kt + 32, cur ^ 1);

    // ---- scores: S[16 x 32] = Q(16x64) @ K^T, from LDS K tile ----
    v16bf b00 = load_frag_b_k(&Ks[cur][0], KLD, hlf, lr);
    v16bf b01 = load_frag_b_k(&Ks[cur][32], KLD, hlf, lr);
    v16bf b10 = load_frag_b_k(&Ks[cur][16 * KLD], KLD, hlf, lr);
    v16bf b11 = load_frag_b_k(&Ks[cur][16 * KLD + 32], KLD, hlf, lr);
    v8f s0 = {}, s1 = {};
    s0 = wmma_bf16(a0, b00, s0);
    s0 = wmma_bf16(a1, b01, s0);
    s1 = wmma_bf16(a0, b10, s1);
    s1 = wmma_bf16(a1, b11, s1);

    float* Sw = sS[w];
#pragma unroll
    for (int r = 0; r < 8; ++r) {
      int m = r + 8 * hlf;
      Sw[m * 33 + lr]      = s0[r] * 0.125f + Bp[(size_t)m * 2048 + kt + lr];
      Sw[m * 33 + 16 + lr] = s1[r] * 0.125f + Bp[(size_t)m * 2048 + kt + 16 + lr];
    }
    lds_fence();   // intra-wave exchange: compiler fence only

    // ---- online softmax: lane owns row lr, half-column hlf ----
    const float* Srow = sS[w] + lr * 33 + hlf * 16;
    float lm = -1e30f;
#pragma unroll
    for (int j = 0; j < 16; ++j) lm = fmaxf(lm, Srow[j]);
    lm = fmaxf(lm, __shfl_xor(lm, 16, 32));
    float mnew = fmaxf(mrow, lm);
    float ps = 0.f;
    __bf16* Prow = sP[w] + lr * 34 + hlf * 16;
#pragma unroll
    for (int j = 0; j < 16; ++j) {
      float p = __expf(Srow[j] - mnew);
      ps += p;
      Prow[j] = (__bf16)p;
    }
    ps += __shfl_xor(ps, 16, 32);
    float alpha = __expf(mrow - mnew);
    lrow = lrow * alpha + ps;
    mrow = mnew;
    if (hlf == 0) sAl[w][lr] = alpha;
    lds_fence();

    // ---- rescale accumulators, then O += P(16x32) @ V(32x64) ----
#pragma unroll
    for (int r = 0; r < 8; ++r) {
      float al = sAl[w][r + 8 * hlf];
#pragma unroll
      for (int t = 0; t < 4; ++t) acc[t][r] *= al;
    }
    v16bf ap = load_frag_a(sP[w], 34, hlf, lr);
#pragma unroll
    for (int t = 0; t < 4; ++t) {
      // B[k][n] = V[kt+k][t*16+n] = Vs[(t*16+n)*VLD + k]
      v16bf bv = load_frag_b_k(&Vs[cur][(t * 16) * VLD], VLD, hlf, lr);
      acc[t] = wmma_bf16(ap, bv, acc[t]);
    }
    cur ^= 1;
  }

  if (hlf == 0) sL[w][lr] = lrow;
  lds_fence();

  // ---- write O / l, concatenating heads: out[b, s, h*64 + e] ----
#pragma unroll
  for (int r = 0; r < 8; ++r) {
    int m = r + 8 * hlf;
    float inv = 1.f / sL[w][m];
    size_t row = (size_t)b * 2048 + (size_t)q0 + m;
#pragma unroll
    for (int t = 0; t < 4; ++t)
      out[row * 1024 + h * 64 + t * 16 + lr] = (__bf16)(acc[t][r] * inv);
  }
}

// ---------------------------------------------------------------------------
// Host-side orchestration
// ---------------------------------------------------------------------------
extern "C" void kernel_launch(void* const* d_in, const int* in_sizes, int n_in,
                              void* d_out, int out_size, void* d_ws,
                              size_t ws_size, hipStream_t stream) {
  (void)in_sizes; (void)n_in; (void)out_size; (void)ws_size;

  const float* x      = (const float*)d_in[0];
  const float* abias  = (const float*)d_in[1];
  const float* ln1_g  = (const float*)d_in[2];
  const float* ln1_b  = (const float*)d_in[3];
  const float* ln2_g  = (const float*)d_in[4];
  const float* ln2_b  = (const float*)d_in[5];
  const float* w_qkv  = (const float*)d_in[6];
  const float* b_qkv  = (const float*)d_in[7];
  const float* w_out  = (const float*)d_in[8];
  const float* b_out  = (const float*)d_in[9];
  const float* w_ff1  = (const float*)d_in[10];
  const float* b_ff1  = (const float*)d_in[11];
  const float* w_ff2  = (const float*)d_in[12];
  const float* b_ff2  = (const float*)d_in[13];

  const int M = 4096;        // B*S
  const int D = 1024, D3 = 3072, DFF = 4096;
  const size_t HSE = (size_t)2 * 16 * 2048 * 64;   // per-tensor Q/K/V elems

  size_t off = 0;
  auto alloc = [&](size_t bytes) -> void* {
    off = (off + 255) & ~(size_t)255;
    void* p = (char*)d_ws + off;
    off += bytes;
    return p;
  };

  __bf16* wqkv_bf = (__bf16*)alloc((size_t)D3 * D * 2);
  __bf16* wout_bf = (__bf16*)alloc((size_t)D * D * 2);
  __bf16* w1_bf   = (__bf16*)alloc((size_t)DFF * D * 2);
  __bf16* w2_bf   = (__bf16*)alloc((size_t)D * DFF * 2);
  __bf16* nx1     = (__bf16*)alloc((size_t)M * D * 2);
  __bf16* qkv_bf  = (__bf16*)alloc((size_t)M * D3 * 2);
  __bf16* Qb      = (__bf16*)alloc(HSE * 2);
  __bf16* Kb      = (__bf16*)alloc(HSE * 2);
  __bf16* Vtb     = (__bf16*)alloc(HSE * 2);
  __bf16* attn_bf = (__bf16*)alloc((size_t)M * D * 2);
  float*  resid1  = (float*)alloc((size_t)M * D * 4);
  __bf16* nx2     = (__bf16*)alloc((size_t)M * D * 2);
  __bf16* h_bf    = (__bf16*)alloc((size_t)M * DFF * 2);

  dim3 blk(256);

  // 1) weights -> bf16
  f2bf_kernel<<<2048, blk, 0, stream>>>(w_qkv, wqkv_bf, (size_t)D3 * D);
  f2bf_kernel<<<2048, blk, 0, stream>>>(w_out, wout_bf, (size_t)D * D);
  f2bf_kernel<<<2048, blk, 0, stream>>>(w_ff1, w1_bf, (size_t)DFF * D);
  f2bf_kernel<<<2048, blk, 0, stream>>>(w_ff2, w2_bf, (size_t)D * DFF);

  // 2) LN1
  ln_kernel<<<M, blk, 0, stream>>>(x, ln1_g, ln1_b, nx1, D);

  // 3) QKV projection: [4096,1024] x [3072,1024]^T
  gemm_kernel<<<dim3(M / 128, D3 / 128), blk, 0, stream>>>(
      nx1, wqkv_bf, b_qkv, nullptr, nullptr, qkv_bf, M, D3, D, 0);

  // 4) split heads (V transposed)
  qkv_split_kernel<<<2048, blk, 0, stream>>>(qkv_bf, Qb, Kb, Vtb);

  // 5) flash attention: 512 blocks, each (b, h, 128 queries)
  attn_kernel<<<512, blk, 0, stream>>>(Qb, Kb, Vtb, abias, attn_bf);

  // 6) out-proj + residual(x) -> resid1 (f32)
  gemm_kernel<<<dim3(M / 128, D / 128), blk, 0, stream>>>(
      attn_bf, wout_bf, b_out, x, resid1, nullptr, M, D, D, 0);

  // 7) LN2
  ln_kernel<<<M, blk, 0, stream>>>(resid1, ln2_g, ln2_b, nx2, D);

  // 8) FFN1 + exact GELU -> h (bf16)
  gemm_kernel<<<dim3(M / 128, DFF / 128), blk, 0, stream>>>(
      nx2, w1_bf, b_ff1, nullptr, nullptr, h_bf, M, DFF, D, 1);

  // 9) FFN2 + residual(resid1) -> d_out (f32)
  gemm_kernel<<<dim3(M / 128, D / 128), blk, 0, stream>>>(
      h_bf, w2_bf, b_ff2, resid1, (float*)d_out, nullptr, M, D, DFF, 0);
}
